// PesudoLabelGenerator_v1_8_ot_rev_3702261809691
// MI455X (gfx1250) — matrix operand order repaired and checked by reading, not compiled, not run on hardware
//
#include <hip/hip_runtime.h>
#include <hip/hip_bf16.h>

#define NSV 8192
#define MSV 8192
#define CF  96
#define NCLS 20

typedef __attribute__((ext_vector_type(16))) __bf16 bf16x16;
typedef __attribute__((ext_vector_type(8)))  __bf16 bf16x8;
typedef __attribute__((ext_vector_type(8)))  float  f32x8;
typedef unsigned long long u64;

// ---- WMMA fragment loaders from pre-normalized bf16 matrices ----
// A: 16x32 bf16, lane(M)=lane%16, half=lane/16; per-lane K runs: [k0+16h..k0+16h+15] (contiguous)
static __device__ __forceinline__ bf16x16 fragA_bf(const __bf16* __restrict__ row, int half, int k0) {
  bf16x8 lo = *(const bf16x8*)(row + k0 + 16 * half);       // 16B aligned
  bf16x8 hi = *(const bf16x8*)(row + k0 + 16 * half + 8);   // 16B aligned
  return __builtin_shufflevector(lo, hi, 0,1,2,3,4,5,6,7,8,9,10,11,12,13,14,15);
}
// B: 32x16 bf16, lane(N)=lane%16; lane-half h holds K=k0+16h..k0+16h+15 (contiguous)
static __device__ __forceinline__ bf16x16 fragB_bf(const __bf16* __restrict__ row, int half, int k0) {
  return *(const bf16x16*)(row + k0 + 16 * half);           // 32B aligned
}

static __device__ __forceinline__ f32x8 wmma_bf16(bf16x16 a, bf16x16 b, f32x8 c) {
  return __builtin_amdgcn_wmma_f32_16x16x32_bf16(false, a, false, b, (short)0, c, false, false);
}

// K[n,m] from cosine (dot of normalized rows) and squared coord distance s:
//   cd = 4*s; support = cd < 100 <=> s < 25
//   K = exp(-(2 - dot - exp(-2s))/0.03)  -> base-2: exp2 with log2(e) folded into constants
static __device__ __forceinline__ float kval(float dot, float dx, float dy, float dz) {
  float s = fmaf(dz, dz, fmaf(dy, dy, dx * dx));
  float g = __builtin_amdgcn_exp2f(s * -2.8853900817779268f);                     // exp(-2s)
  float e = __builtin_amdgcn_exp2f(fmaf(dot + g, 48.08983469629878f, -96.17966939259756f));
  return (s < 25.0f) ? e : 0.0f;
}

// ---------------- init ----------------
__global__ void k_init(u64* packed, float* rowsum, unsigned* flag, int* owner, float* trust) {
  int i = blockIdx.x * blockDim.x + threadIdx.x;
  if (i < MSV) packed[i] = 0ull;
  if (i < NSV) { rowsum[i] = 0.0f; trust[i] = 0.0f; owner[i] = -1; }
  if (i == 0) *flag = 0u;
}

// ---------------- pose: diff = inv(P1) @ P0 ----------------
__global__ void k_setup(const float* __restrict__ posses, float* __restrict__ diffm) {
  if (threadIdx.x != 0 || blockIdx.x != 0) return;
  float a[4][8];
  for (int i = 0; i < 4; ++i)
    for (int j = 0; j < 4; ++j) { a[i][j] = posses[16 + i * 4 + j]; a[i][4 + j] = (i == j) ? 1.0f : 0.0f; }
  for (int c = 0; c < 4; ++c) {
    int p = c;
    for (int r = c + 1; r < 4; ++r) if (fabsf(a[r][c]) > fabsf(a[p][c])) p = r;
    if (p != c) for (int j = 0; j < 8; ++j) { float t = a[c][j]; a[c][j] = a[p][j]; a[p][j] = t; }
    float inv = 1.0f / a[c][c];
    for (int j = 0; j < 8; ++j) a[c][j] *= inv;
    for (int r = 0; r < 4; ++r) if (r != c) {
      float f = a[r][c];
      for (int j = 0; j < 8; ++j) a[r][j] -= f * a[c][j];
    }
  }
  for (int i = 0; i < 4; ++i)
    for (int j = 0; j < 4; ++j) {
      float s = 0.0f;
      for (int k = 0; k < 4; ++k) s += a[i][4 + k] * posses[k * 4 + j];
      diffm[i * 4 + j] = s;
    }
}

// ------- normalize features to bf16, aligned coords, per-n side data -------
__global__ void k_prep(const float* __restrict__ mean, const float* __restrict__ sur,
                       const float* __restrict__ coords, const float* __restrict__ ori,
                       const float* __restrict__ diffm,
                       __bf16* __restrict__ meanbf, __bf16* __restrict__ surbf,
                       float4* __restrict__ alsur4, float4* __restrict__ ndata4) {
  int i = blockIdx.x * blockDim.x + threadIdx.x;
  if (i < NSV) {
    const float* row = mean + (size_t)i * CF;
    float s = 0.0f;
    for (int k = 0; k < CF; ++k) s += row[k] * row[k];
    float rs = (s > 0.0f) ? rsqrtf(s) : 0.0f;   // norms ~sqrt(96); 1e-8 clip never binds
    __bf16* orow = meanbf + (size_t)i * CF;
    for (int k8 = 0; k8 < CF; k8 += 8) {
      bf16x8 v;
#pragma unroll
      for (int j = 0; j < 8; ++j) v[j] = (__bf16)(row[k8 + j] * rs);
      *(bf16x8*)(orow + k8) = v;
    }
    ndata4[i] = make_float4(ori[3 * i + 0], ori[3 * i + 1], ori[3 * i + 2], 0.0f);
  }
  if (i < MSV) {
    const float* row = sur + (size_t)i * CF;
    float s = 0.0f;
    for (int k = 0; k < CF; ++k) s += row[k] * row[k];
    float rs = (s > 0.0f) ? rsqrtf(s) : 0.0f;
    __bf16* orow = surbf + (size_t)i * CF;
    for (int k8 = 0; k8 < CF; k8 += 8) {
      bf16x8 v;
#pragma unroll
      for (int j = 0; j < 8; ++j) v[j] = (__bf16)(row[k8 + j] * rs);
      *(bf16x8*)(orow + k8) = v;
    }
    float x = coords[3 * i + 0], y = coords[3 * i + 1], z = coords[3 * i + 2];
    float al[3];
    for (int r = 0; r < 3; ++r)
      al[r] = diffm[r * 4 + 0] * x + diffm[r * 4 + 1] * y + diffm[r * 4 + 2] * z + diffm[r * 4 + 3];
    alsur4[i] = make_float4(al[0], al[1], al[2], 0.0f);
  }
}

// ---------------- pass A: rowsum[n] = sum_m K[n,m] ----------------
__global__ void __launch_bounds__(256) k_passA(
    const __bf16* __restrict__ meanbf, const __bf16* __restrict__ surbf,
    const float4* __restrict__ alsur4, const float4* __restrict__ ndata4,
    float* __restrict__ rowsum) {
  const int wid = threadIdx.x >> 5, lane = threadIdx.x & 31;
  const int half = lane >> 4, lr = lane & 15;
  const int n0 = blockIdx.x * 128 + wid * 16;
  const int mBase = blockIdx.y * 1024;

  const __bf16* arow = meanbf + (size_t)(n0 + lr) * CF;
  bf16x16 A0 = fragA_bf(arow, half, 0);
  bf16x16 A1 = fragA_bf(arow, half, 32);
  bf16x16 A2 = fragA_bf(arow, half, 64);

  float ox[8], oy[8], oz[8], racc[8];
#pragma unroll
  for (int r = 0; r < 8; ++r) {
    float4 nd = ndata4[n0 + r + 8 * half];
    ox[r] = nd.x; oy[r] = nd.y; oz[r] = nd.z;
    racc[r] = 0.0f;
  }

  // software pipeline: F* holds tile ct, G* prefetches tile ct+1
  int m = mBase + lr;
  const __bf16* brow = surbf + (size_t)m * CF;
  bf16x16 F0 = fragB_bf(brow, half, 0);
  bf16x16 F1 = fragB_bf(brow, half, 32);
  bf16x16 F2 = fragB_bf(brow, half, 64);
  float4 md = alsur4[m];

  for (int ct = 0; ct < 64; ++ct) {
    const int m2 = (ct < 63) ? m + 16 : m;            // branchless, stays in-bounds
    const __bf16* nrow = surbf + (size_t)m2 * CF;
    bf16x16 G0 = fragB_bf(nrow, half, 0);
    bf16x16 G1 = fragB_bf(nrow, half, 32);
    bf16x16 G2 = fragB_bf(nrow, half, 64);
    float4 md2 = alsur4[m2];

    f32x8 acc = (f32x8)(0.0f);
    acc = wmma_bf16(A0, F0, acc);
    acc = wmma_bf16(A1, F1, acc);
    acc = wmma_bf16(A2, F2, acc);
#pragma unroll
    for (int r = 0; r < 8; ++r)
      racc[r] += kval(acc[r], ox[r] - md.x, oy[r] - md.y, oz[r] - md.z);

    F0 = G0; F1 = G1; F2 = G2; md = md2; m += 16;
  }
#pragma unroll
  for (int r = 0; r < 8; ++r) {
    float v = racc[r];
    v += __shfl_xor(v, 1, 32);
    v += __shfl_xor(v, 2, 32);
    v += __shfl_xor(v, 4, 32);
    v += __shfl_xor(v, 8, 32);
    if (lr == 0) atomicAdd(&rowsum[n0 + r + 8 * half], v);
  }
}

// ---------------- b vector -> ndata4.w ----------------
__global__ void k_bvec(const float* __restrict__ rowsum, float4* __restrict__ ndata4) {
  int n = blockIdx.x * blockDim.x + threadIdx.x;
  if (n < NSV) ndata4[n].w = (1.0f / NSV) / ((1.0f / MSV) * rowsum[n] + 1e-16f);
}

// ---------------- pass B: sm_index[m] = argmax_n K[n,m]*b[n] ----------------
__global__ void __launch_bounds__(256) k_passB(
    const __bf16* __restrict__ meanbf, const __bf16* __restrict__ surbf,
    const float4* __restrict__ alsur4, const float4* __restrict__ ndata4,
    u64* __restrict__ packed) {
  const int wid = threadIdx.x >> 5, lane = threadIdx.x & 31;
  const int half = lane >> 4, lr = lane & 15;
  const int m = blockIdx.x * 128 + wid * 16 + lr;   // this lane's fixed column
  const int nBase = blockIdx.y * 1024;

  const __bf16* brow = surbf + (size_t)m * CF;
  bf16x16 B0 = fragB_bf(brow, half, 0);
  bf16x16 B1 = fragB_bf(brow, half, 32);
  bf16x16 B2 = fragB_bf(brow, half, 64);
  const float4 md = alsur4[m];

  // software pipeline on the A operand (rows)
  int nrow0 = nBase + lr;
  const __bf16* arow = meanbf + (size_t)nrow0 * CF;
  bf16x16 F0 = fragA_bf(arow, half, 0);
  bf16x16 F1 = fragA_bf(arow, half, 32);
  bf16x16 F2 = fragA_bf(arow, half, 64);

  float bestV = -1.0f; int bestN = 0;
  for (int rt = 0; rt < 64; ++rt) {
    const int n0 = nBase + rt * 16;
    const int nr2 = (rt < 63) ? nrow0 + 16 : nrow0;   // branchless, in-bounds
    const __bf16* nxt = meanbf + (size_t)nr2 * CF;
    bf16x16 G0 = fragA_bf(nxt, half, 0);
    bf16x16 G1 = fragA_bf(nxt, half, 32);
    bf16x16 G2 = fragA_bf(nxt, half, 64);

    // issue side-data loads before the WMMAs so their latency overlaps
    float4 nd[8];
#pragma unroll
    for (int r = 0; r < 8; ++r) nd[r] = ndata4[n0 + r + 8 * half];

    f32x8 acc = (f32x8)(0.0f);
    acc = wmma_bf16(F0, B0, acc);
    acc = wmma_bf16(F1, B1, acc);
    acc = wmma_bf16(F2, B2, acc);
#pragma unroll
    for (int r = 0; r < 8; ++r) {
      int n = n0 + r + 8 * half;
      float v = kval(acc[r], nd[r].x - md.x, nd[r].y - md.y, nd[r].z - md.z) * nd[r].w;
      if (v > bestV) { bestV = v; bestN = n; }   // n increases -> first-occurrence kept
    }
    F0 = G0; F1 = G1; F2 = G2; nrow0 = nr2;
  }
  float oV = __shfl_xor(bestV, 16, 32);
  int   oN = __shfl_xor(bestN, 16, 32);
  if (oV > bestV || (oV == bestV && oN < bestN)) { bestV = oV; bestN = oN; }
  if (half == 0) {
    u64 enc = ((u64)__float_as_uint(bestV) << 32) | (u64)(0xFFFFFFFFu - (unsigned)bestN);
    atomicMax(&packed[m], enc);   // val>=0 -> bit-monotone; ties prefer smaller n
  }
}

// ---------------- softmax rows of sv_prob ----------------
__global__ void k_softmax(const float* __restrict__ svprob, float* __restrict__ probs) {
  int n = blockIdx.x * blockDim.x + threadIdx.x;
  if (n >= NSV) return;
  const float* row = svprob + (size_t)n * NCLS;
  float mx = row[0];
  for (int j = 1; j < NCLS; ++j) mx = fmaxf(mx, row[j]);
  float e[NCLS]; float s = 0.0f;
  for (int j = 0; j < NCLS; ++j) { e[j] = __expf(row[j] - mx); s += e[j]; }
  float inv = 1.0f / s;
  float* orow = probs + (size_t)n * NCLS;
  for (int j = 0; j < NCLS; ++j) orow[j] = e[j] * inv;
}

// ---------------- conf + any(conf>0.1) + emit sm_index ----------------
__global__ void k_conf(const u64* __restrict__ packed, const int* __restrict__ gt,
                       const float* __restrict__ probs, float* __restrict__ smout,
                       float* __restrict__ conf, unsigned* __restrict__ flag,
                       const int* __restrict__ lgp) {
  int m = blockIdx.x * blockDim.x + threadIdx.x;
  if (m >= MSV) return;
  unsigned sm = 0xFFFFFFFFu - (unsigned)(packed[m] & 0xFFFFFFFFull);
  smout[m] = (float)sm;
  if (m >= *lgp) return;
  int c = gt[m]; if (c < 0 || c >= NCLS) c = 0;  // one_hot OOB -> argmax 0
  float cf = probs[(size_t)sm * NCLS + c];
  conf[m] = cf;
  if (cf > 0.1f) atomicOr(flag, 1u);
}

// ------- scatter phase 1: deterministic owner (largest m wins) -------
__global__ void k_scatter1(const u64* __restrict__ packed, const float* __restrict__ conf,
                           const unsigned* __restrict__ flag, int* __restrict__ owner,
                           const int* __restrict__ lgp) {
  int m = blockIdx.x * blockDim.x + threadIdx.x;
  if (m >= MSV || m >= *lgp) return;
  bool tmp = (*flag) ? (conf[m] > 0.1f) : (conf[m] > 0.0f);
  if (!tmp) return;
  unsigned sm = 0xFFFFFFFFu - (unsigned)(packed[m] & 0xFFFFFFFFull);
  atomicMax(&owner[sm], m);
}

// ------- scatter phase 2: write one-hot rows + trust -------
__global__ void k_scatter2(const int* __restrict__ owner, const int* __restrict__ gt,
                           float* __restrict__ probs, float* __restrict__ trust) {
  int n = blockIdx.x * blockDim.x + threadIdx.x;
  if (n >= NSV) return;
  int m = owner[n];
  if (m < 0) return;
  int c = gt[m]; if (c < 0 || c >= NCLS) c = -1;
  float* orow = probs + (size_t)n * NCLS;
  for (int j = 0; j < NCLS; ++j) orow[j] = (j == c) ? 1.0f : 0.0f;
  trust[n] = 1.0f;
}

// ---------------- pre = row argmax of updated probs ----------------
__global__ void k_pre(const float* __restrict__ probs, float* __restrict__ pre) {
  int n = blockIdx.x * blockDim.x + threadIdx.x;
  if (n >= NSV) return;
  const float* row = probs + (size_t)n * NCLS;
  float bv = row[0]; int bi = 0;
  for (int j = 1; j < NCLS; ++j) if (row[j] > bv) { bv = row[j]; bi = j; }
  pre[n] = (float)bi;
}

extern "C" void kernel_launch(void* const* d_in, const int* in_sizes, int n_in,
                              void* d_out, int out_size, void* d_ws, size_t ws_size,
                              hipStream_t stream) {
  (void)in_sizes; (void)n_in; (void)out_size; (void)ws_size;
  const float* sur     = (const float*)d_in[0];  // (M,96)
  const float* scoords = (const float*)d_in[1];  // (M,3)
  const int*   gt      = (const int*)d_in[2];    // (M,)
  const float* svprob  = (const float*)d_in[3];  // (N,20)
  const float* mean    = (const float*)d_in[4];  // (N,96)
  const float* ori     = (const float*)d_in[5];  // (N,3)
  const float* posses  = (const float*)d_in[6];  // (2,4,4)
  const int*   lgp     = (const int*)d_in[8];    // scalar

  float* out   = (float*)d_out;
  float* trust = out;                            // N
  float* probs = out + NSV;                      // N*20
  float* pre   = probs + (size_t)NSV * NCLS;     // N
  float* smout = pre + NSV;                      // M

  char* ws = (char*)d_ws;
  u64*    packed = (u64*)ws;            ws += (size_t)MSV * 8;        // 32B-aligned offsets
  __bf16* meanbf = (__bf16*)ws;         ws += (size_t)NSV * CF * 2;
  __bf16* surbf  = (__bf16*)ws;         ws += (size_t)MSV * CF * 2;
  float4* alsur4 = (float4*)ws;         ws += (size_t)MSV * 16;
  float4* ndata4 = (float4*)ws;         ws += (size_t)NSV * 16;
  float*  diffm  = (float*)ws;          ws += 64;
  float*  rowsum = (float*)ws;          ws += (size_t)NSV * 4;
  float*  conf   = (float*)ws;          ws += (size_t)MSV * 4;
  int*    owner  = (int*)ws;            ws += (size_t)NSV * 4;
  unsigned* flag = (unsigned*)ws;

  const int T = 256;
  k_init<<<(MSV + T - 1) / T, T, 0, stream>>>(packed, rowsum, flag, owner, trust);
  k_setup<<<1, 32, 0, stream>>>(posses, diffm);
  k_prep<<<(MSV + T - 1) / T, T, 0, stream>>>(mean, sur, scoords, ori, diffm,
                                              meanbf, surbf, alsur4, ndata4);
  k_passA<<<dim3(NSV / 128, MSV / 1024), T, 0, stream>>>(meanbf, surbf, alsur4, ndata4, rowsum);
  k_bvec<<<(NSV + T - 1) / T, T, 0, stream>>>(rowsum, ndata4);
  k_passB<<<dim3(MSV / 128, NSV / 1024), T, 0, stream>>>(meanbf, surbf, alsur4, ndata4, packed);
  k_softmax<<<(NSV + T - 1) / T, T, 0, stream>>>(svprob, probs);
  k_conf<<<(MSV + T - 1) / T, T, 0, stream>>>(packed, gt, probs, smout, conf, flag, lgp);
  k_scatter1<<<(MSV + T - 1) / T, T, 0, stream>>>(packed, conf, flag, owner, lgp);
  k_scatter2<<<(NSV + T - 1) / T, T, 0, stream>>>(owner, gt, probs, trust);
  k_pre<<<(NSV + T - 1) / T, T, 0, stream>>>(probs, pre);
}
